// DynamicsPredictor_50672024158795
// MI455X (gfx1250) — matrix operand order repaired and checked by reading, not compiled, not run on hardware
//
#include <hip/hip_runtime.h>
#include <hip/hip_bf16.h>

#define B_     2
#define NP_    1400
#define NN_    1500
#define NRL_   24000
#define NI_    8
#define NF_    128
#define KMAX_  384

typedef unsigned short u16;
typedef __attribute__((ext_vector_type(16))) __bf16       v16bf;
typedef __attribute__((ext_vector_type(8)))  float        v8f;
typedef __attribute__((ext_vector_type(4)))  unsigned int u32x4;

__device__ __forceinline__ u16 f2bf(float f) {
    union { float f; unsigned int u; } c; c.f = f;
    unsigned int r = c.u + 0x7fffu + ((c.u >> 16) & 1u);
    return (u16)(r >> 16);
}

// ---------------------------------------------------------------------------
// Generic fused bf16 WMMA GEMM:  D[M x ncols] = act(A[M x K] * W[K x 128] + bias (+res))
//  - W is pre-transposed: Wt[n][Kpad]  (L2-resident, B frags read directly, no LDS)
//  - A panel (16 x K, K<=384) staged in LDS once; single barrier per block
//  - optional gather-built A (propagation step) and atomic scatter-add epilogue
// ---------------------------------------------------------------------------
struct GemmArgs {
    const u16*  A;        // bf16 A, row stride Ak (null => gather mode)
    int         Ak;
    const u16*  W;        // bf16 weights TRANSPOSED: [128][K] (zero padded)
    const float* bias;    // 128 (zero padded)
    int         M, K;     // K multiple of 32, K <= 384
    // gather mode: row r = [g_rel[r](128) | g_eff[idxR[r]](128) | g_eff[idxS[r]](128)]
    const u16*  g_rel; const u16* g_eff; const int* g_idxR; const int* g_idxS;
    float*      outF; int outF_stride; int outF_ncols;   // f32 out (nullable)
    u16*        outB;                                    // bf16 out, stride 128 (nullable)
    const float* residual;                               // stride 128, added pre-relu
    int         relu;
    float*      scatter_out; const int* scatter_idx;     // atomicAdd epilogue (nullable)
};

__launch_bounds__(256)
__global__ void gemm_bf16_kernel(GemmArgs ga) {
    __shared__ __align__(16) u16 As[16 * KMAX_];   // A panel, [m][k], stride K
    const int tid  = threadIdx.x;
    const int wave = tid >> 5;
    const int lane = tid & 31;
    const int row0 = blockIdx.x * 16;
    const int K    = ga.K;
    const int halfK = K >> 1;                      // u32 slots per row

    // ---- stage full A panel: 16 x K bf16 (8*K u32 slots) ----
    for (int q = tid; q < 8 * K; q += 256) {
        const int m  = q / halfK;
        const int kk = (q - m * halfK) * 2;
        const int r  = row0 + m;
        unsigned int val = 0u;
        if (r < ga.M) {
            if (ga.g_rel) {
                const u16* src; int col;
                if (kk < 128)      { src = ga.g_rel + (size_t)r * 128;             col = kk;       }
                else if (kk < 256) { src = ga.g_eff + (size_t)ga.g_idxR[r] * 128;  col = kk - 128; }
                else               { src = ga.g_eff + (size_t)ga.g_idxS[r] * 128;  col = kk - 256; }
                val = *(const unsigned int*)(src + col);
            } else {
                val = *(const unsigned int*)(ga.A + (size_t)r * ga.Ak + kk);
            }
        }
        *(unsigned int*)(As + m * K + kk) = val;
    }
    __syncthreads();

    // ---- K loop: pure ds_load_b128 (A) + global_load_b128 (Wt from L2) + wmma ----
    const int half = lane >> 4;
    const int mrow = lane & 15;
    const int gcol = wave * 16 + (lane & 15);
    const u16* wt  = ga.W + (size_t)gcol * K;      // transposed weight row for this col
    v8f acc = {};
    for (int k0 = 0; k0 < K; k0 += 32) {
        union { u32x4 q[2]; v16bf v; } af, bf;
        af.q[0] = *(const u32x4*)(As + mrow * K + k0 + half * 8);       // K = k0+8*half+0..7
        af.q[1] = *(const u32x4*)(As + mrow * K + k0 + 16 + half * 8);  // K = k0+16+8*half+0..7
        const u16* bp = wt + k0 + half * 8;                             // K = k0+8*half+0..7, k0+16+8*half+0..7
        bf.q[0] = *(const u32x4*)(bp);
        bf.q[1] = *(const u32x4*)(bp + 16);
        acc = __builtin_amdgcn_wmma_f32_16x16x32_bf16(false, af.v, false, bf.v,
                                                      (short)0, acc, false, false);
    }

    // ---- epilogue ----
    const int mbase = (lane >> 4) * 8;
    const float bv  = ga.bias[gcol];
    for (int e = 0; e < 8; ++e) {
        const int r = row0 + mbase + e;
        if (r >= ga.M) continue;
        float v = acc[e] + bv;
        if (ga.residual) v += ga.residual[(size_t)r * 128 + gcol];
        if (ga.relu) v = v > 0.f ? v : 0.f;
        if (ga.scatter_out)
            atomicAdd(ga.scatter_out + (size_t)ga.scatter_idx[r] * 128 + gcol, v);
        if (ga.outB)
            ga.outB[(size_t)r * 128 + gcol] = f2bf(v);
        if (ga.outF && gcol < ga.outF_ncols)
            ga.outF[(size_t)r * ga.outF_stride + gcol] = v;
    }
}

// ---------------------------------------------------------------------------
// One-hot (rows of length 1500) -> global node index.  Wave per row; the only
// kernel touching the 288MB Rr/Rs matrices (streaming, float4 + prefetch).
// ---------------------------------------------------------------------------
__launch_bounds__(256)
__global__ void onehot_idx_kernel(const float* __restrict__ R, int* __restrict__ idx,
                                  int nrows) {
    const int row  = blockIdx.x * 8 + (threadIdx.x >> 5);
    const int lane = threadIdx.x & 31;
    if (row >= nrows) return;
    const float4* p = (const float4*)(R + (size_t)row * NN_);
    int best = 0;
    for (int i = lane; i < NN_ / 4; i += 32) {
        __builtin_prefetch(p + i + 64, 0, 1);     // global_prefetch_b8, next chunk
        const float4 v = p[i];
        if (v.x > 0.5f) best = 4 * i;
        if (v.y > 0.5f) best = 4 * i + 1;
        if (v.z > 0.5f) best = 4 * i + 2;
        if (v.w > 0.5f) best = 4 * i + 3;
    }
    for (int off = 16; off; off >>= 1) {
        const int o = __shfl_xor(best, off, 32);
        best = best > o ? best : o;
    }
    if (lane == 0) idx[row] = (row / NRL_) * NN_ + best;   // globalized node row
}

// Per-instance center: sum(p_instance * sp) / (count + eps)
__launch_bounds__(256)
__global__ void inst_center_kernel(const float* __restrict__ state,
                                   const float* __restrict__ p_instance,
                                   float* __restrict__ center) {
    __shared__ float sm[NI_ * 4];
    const int b = blockIdx.x;
    if (threadIdx.x < NI_ * 4) sm[threadIdx.x] = 0.f;
    __syncthreads();
    float loc[NI_][4];
    for (int i = 0; i < NI_; ++i) loc[i][0] = loc[i][1] = loc[i][2] = loc[i][3] = 0.f;
    for (int p = threadIdx.x; p < NP_; p += 256) {
        const float* sp = state + ((size_t)b * NN_ + p) * 3;
        const float* pi = p_instance + ((size_t)b * NP_ + p) * NI_;
        const float s0 = sp[0], s1 = sp[1], s2 = sp[2];
        for (int i = 0; i < NI_; ++i) {
            const float w = pi[i];
            loc[i][0] += w * s0; loc[i][1] += w * s1; loc[i][2] += w * s2; loc[i][3] += w;
        }
    }
    for (int i = 0; i < NI_; ++i)
        for (int d = 0; d < 4; ++d) atomicAdd(&sm[i * 4 + d], loc[i][d]);
    __syncthreads();
    if (threadIdx.x < NI_) {
        const int i = threadIdx.x;
        const float inv = 1.f / (sm[i * 4 + 3] + 1e-6f);
        center[(b * NI_ + i) * 3 + 0] = sm[i * 4 + 0] * inv;
        center[(b * NI_ + i) * 3 + 1] = sm[i * 4 + 1] * inv;
        center[(b * NI_ + i) * 3 + 2] = sm[i * 4 + 2] * inv;
    }
}

// p_inputs = [attrs(3), snt(3), offset(3)] -> bf16 rows padded to 32
__launch_bounds__(256)
__global__ void p_inputs_kernel(const float* __restrict__ state, const float* __restrict__ attrs,
                                const float* __restrict__ p_instance, const float* __restrict__ p_rigid,
                                const float* __restrict__ center, u16* __restrict__ pin) {
    const int r = blockIdx.x * 256 + threadIdx.x;
    if (r >= B_ * NN_) return;
    const int b = r / NN_, n = r % NN_;
    const float* sp = state + (size_t)r * 3;
    const float* at = attrs + (size_t)r * 3;
    float off0 = 0.f, off1 = 0.f, off2 = 0.f;
    if (n < NP_) {
        const float* pi = p_instance + ((size_t)b * NP_ + n) * NI_;
        float prpp = 0.f, c0 = 0.f, c1 = 0.f, c2 = 0.f;
        for (int i = 0; i < NI_; ++i) {
            const float w = pi[i];
            prpp += w * p_rigid[b * NI_ + i];
            c0 += w * center[(b * NI_ + i) * 3 + 0];
            c1 += w * center[(b * NI_ + i) * 3 + 1];
            c2 += w * center[(b * NI_ + i) * 3 + 2];
        }
        off0 = prpp * (sp[0] - c0); off1 = prpp * (sp[1] - c1); off2 = prpp * (sp[2] - c2);
    }
    u16* o = pin + (size_t)r * 32;
    o[0] = f2bf(at[0]); o[1] = f2bf(at[1]); o[2] = f2bf(at[2]);
    o[3] = f2bf(sp[0]); o[4] = f2bf(sp[1]); o[5] = f2bf(sp[2]);
    o[6] = f2bf(off0);  o[7] = f2bf(off1);  o[8] = f2bf(off2);
    for (int k = 9; k < 32; ++k) o[k] = 0;
}

// rel_inputs = [p_r(9), p_s(9), a_r(3), a_s(3), gdiff(1), pos_diff(3)] -> bf16, padded 32
__launch_bounds__(256)
__global__ void rel_inputs_kernel(const int* __restrict__ idxR, const int* __restrict__ idxS,
                                  const u16* __restrict__ pin, const float* __restrict__ attrs,
                                  const float* __restrict__ state, const float* __restrict__ p_instance,
                                  u16* __restrict__ relin) {
    const int r = blockIdx.x * 256 + threadIdx.x;
    if (r >= B_ * NRL_) return;
    const int ir = idxR[r], is = idxS[r];
    u16* o = relin + (size_t)r * 32;
    const u16* pr = pin + (size_t)ir * 32;
    const u16* ps = pin + (size_t)is * 32;
    for (int k = 0; k < 9; ++k) { o[k] = pr[k]; o[9 + k] = ps[k]; }
    const float* ar = attrs + (size_t)ir * 3;
    const float* as = attrs + (size_t)is * 3;
    o[18] = f2bf(ar[0]); o[19] = f2bf(ar[1]); o[20] = f2bf(ar[2]);
    o[21] = f2bf(as[0]); o[22] = f2bf(as[1]); o[23] = f2bf(as[2]);
    const int br = ir / NN_, prn = ir % NN_;
    const int bs = is / NN_, psn = is % NN_;
    float gd = 0.f;
    for (int i = 0; i < NI_; ++i) {
        const float gr = (prn < NP_) ? p_instance[((size_t)br * NP_ + prn) * NI_ + i] : 0.f;
        const float gs = (psn < NP_) ? p_instance[((size_t)bs * NP_ + psn) * NI_ + i] : 0.f;
        gd += fabsf(gr - gs);
    }
    o[24] = f2bf(gd);
    const float* sr = state + (size_t)ir * 3;
    const float* ss = state + (size_t)is * 3;
    o[25] = f2bf(sr[0] - ss[0]); o[26] = f2bf(sr[1] - ss[1]); o[27] = f2bf(sr[2] - ss[2]);
    o[28] = o[29] = o[30] = o[31] = 0;
}

__launch_bounds__(256)
__global__ void zero_kernel(float* __restrict__ p, int n) {
    const int t = blockIdx.x * 256 + threadIdx.x;
    if (t < n) p[t] = 0.f;
}

// concat [particle_encode(bf16) | agg(f32->bf16)] -> 3000 x 256 bf16
__launch_bounds__(256)
__global__ void concat_pp_kernel(const u16* __restrict__ peb, const float* __restrict__ agg,
                                 u16* __restrict__ out) {
    const int t = blockIdx.x * 256 + threadIdx.x;
    if (t >= B_ * NN_ * 256) return;
    const int r = t >> 8, k = t & 255;
    out[t] = (k < 128) ? peb[(size_t)r * 128 + k] : f2bf(agg[(size_t)r * 128 + (k - 128)]);
}

// extract eff[:, :NP_] -> 2800 x 128 bf16
__launch_bounds__(256)
__global__ void extract_h_kernel(const u16* __restrict__ effb, u16* __restrict__ h0) {
    const int t = blockIdx.x * 256 + threadIdx.x;
    if (t >= B_ * NP_ * 128) return;
    const int r = t >> 7, k = t & 127;
    const int b = r / NP_, p = r % NP_;
    h0[t] = effb[((size_t)(b * NN_ + p)) * 128 + k];
}

// weights f32 KxNcols -> bf16 TRANSPOSED [128][Kpad] (zero padded); bias -> 128 padded
__launch_bounds__(256)
__global__ void convw_kernel(const float* __restrict__ w, const float* __restrict__ b,
                             int K, int Ncols, int Kpad, u16* __restrict__ wOut,
                             float* __restrict__ bOut) {
    const int t = blockIdx.x * 256 + threadIdx.x;
    if (t < Kpad * 128) {
        const int n = t / Kpad;           // output column 0..127
        const int k = t - n * Kpad;       // K index
        wOut[t] = (k < K && n < Ncols) ? f2bf(w[(size_t)k * Ncols + n]) : (u16)0;
    }
    if (t < 128) bOut[t] = (t < Ncols) ? b[t] : 0.f;
}

// ---------------------------------------------------------------------------
extern "C" void kernel_launch(void* const* d_in, const int* in_sizes, int n_in,
                              void* d_out, int out_size, void* d_ws, size_t ws_size,
                              hipStream_t stream) {
    (void)in_sizes; (void)n_in; (void)out_size; (void)ws_size;
    const float* state      = (const float*)d_in[0];
    const float* attrs      = (const float*)d_in[1];
    const float* Rr         = (const float*)d_in[2];
    const float* Rs         = (const float*)d_in[3];
    const float* p_instance = (const float*)d_in[4];
    const float* p_rigid    = (const float*)d_in[5];
    // params flattened in dict order: pe(3x w,b), re(3x w,b), rp(w,b), pp(w,b), nr(3x w,b)
    const float* W_in[11] = { (const float*)d_in[6],  (const float*)d_in[8],  (const float*)d_in[10],
                              (const float*)d_in[12], (const float*)d_in[14], (const float*)d_in[16],
                              (const float*)d_in[18], (const float*)d_in[20],
                              (const float*)d_in[22], (const float*)d_in[24], (const float*)d_in[26] };
    const float* Bi_in[11] = { (const float*)d_in[7],  (const float*)d_in[9],  (const float*)d_in[11],
                               (const float*)d_in[13], (const float*)d_in[15], (const float*)d_in[17],
                               (const float*)d_in[19], (const float*)d_in[21],
                               (const float*)d_in[23], (const float*)d_in[25], (const float*)d_in[27] };
    const int  Wk[11]  = { 9, 128, 128, 28, 128, 128, 384, 256, 128, 128, 128 };
    const int  Wn[11]  = { 128, 128, 128, 128, 128, 128, 128, 128, 128, 128, 3 };
    const int  Wkp[11] = { 32, 128, 128, 32, 128, 128, 384, 256, 128, 128, 128 };

    char* ws = (char*)d_ws;
    size_t off = 0;
    auto alloc = [&](size_t bytes) -> void* {
        void* p = ws + off; off += (bytes + 255) & ~(size_t)255; return p;
    };
    int*   idxR   = (int*)  alloc((size_t)B_ * NRL_ * 4);
    int*   idxS   = (int*)  alloc((size_t)B_ * NRL_ * 4);
    float* center = (float*)alloc((size_t)B_ * NI_ * 3 * 4);
    u16*   pin    = (u16*)  alloc((size_t)B_ * NN_ * 32 * 2);
    u16*   relin  = (u16*)  alloc((size_t)B_ * NRL_ * 32 * 2);
    u16*   tmpA   = (u16*)  alloc((size_t)B_ * NRL_ * 128 * 2);
    u16*   tmpB   = (u16*)  alloc((size_t)B_ * NRL_ * 128 * 2);
    u16*   relenc = (u16*)  alloc((size_t)B_ * NRL_ * 128 * 2);
    u16*   peb    = (u16*)  alloc((size_t)B_ * NN_ * 128 * 2);
    float* E0     = (float*)alloc((size_t)B_ * NN_ * 128 * 4);
    float* E1     = (float*)alloc((size_t)B_ * NN_ * 128 * 4);
    u16*   eb0    = (u16*)  alloc((size_t)B_ * NN_ * 128 * 2);
    u16*   eb1    = (u16*)  alloc((size_t)B_ * NN_ * 128 * 2);
    float* agg    = (float*)alloc((size_t)B_ * NN_ * 128 * 4);
    u16*   cpp    = (u16*)  alloc((size_t)B_ * NN_ * 256 * 2);
    u16*   Wb[11]; float* Bb[11];
    for (int i = 0; i < 11; ++i) {
        Wb[i] = (u16*)  alloc((size_t)Wkp[i] * 128 * 2);
        Bb[i] = (float*)alloc(128 * 4);
    }

    // weight conversion + transpose (cheap, deterministic, L2-resident afterwards)
    for (int i = 0; i < 11; ++i)
        convw_kernel<<<(Wkp[i] * 128 + 255) / 256, 256, 0, stream>>>(
            W_in[i], Bi_in[i], Wk[i], Wn[i], Wkp[i], Wb[i], Bb[i]);

    // one-hot -> index (the 576MB streaming pass)
    const int nrel = B_ * NRL_;
    onehot_idx_kernel<<<(nrel + 7) / 8, 256, 0, stream>>>(Rr, idxR, nrel);
    onehot_idx_kernel<<<(nrel + 7) / 8, 256, 0, stream>>>(Rs, idxS, nrel);

    inst_center_kernel<<<B_, 256, 0, stream>>>(state, p_instance, center);
    p_inputs_kernel<<<(B_ * NN_ + 255) / 256, 256, 0, stream>>>(
        state, attrs, p_instance, p_rigid, center, pin);

    auto gemm = [&](const u16* A, int Ak, int wi, int M, int K,
                    float* outF, int outF_stride, int outF_ncols, u16* outB,
                    const float* residual, int relu,
                    const u16* grel, const u16* geff, const int* gR, const int* gS,
                    float* scat, const int* scatIdx) {
        GemmArgs ga;
        ga.A = A; ga.Ak = Ak; ga.W = Wb[wi]; ga.bias = Bb[wi]; ga.M = M; ga.K = K;
        ga.g_rel = grel; ga.g_eff = geff; ga.g_idxR = gR; ga.g_idxS = gS;
        ga.outF = outF; ga.outF_stride = outF_stride; ga.outF_ncols = outF_ncols;
        ga.outB = outB; ga.residual = residual; ga.relu = relu;
        ga.scatter_out = scat; ga.scatter_idx = scatIdx;
        gemm_bf16_kernel<<<(M + 15) / 16, 256, 0, stream>>>(ga);
    };

    const int Mn = B_ * NN_;       // 3000
    const int Mr = B_ * NRL_;      // 48000
    const int Mp = B_ * NP_;       // 2800

    // particle encoder: 9->128->128->128 (relu), out f32 (eff init) + bf16
    gemm(pin,  32, 0, Mn,  32, nullptr, 0, 0, tmpA, nullptr, 1, 0, 0, 0, 0, nullptr, nullptr);
    gemm(tmpA, 128, 1, Mn, 128, nullptr, 0, 0, tmpB, nullptr, 1, 0, 0, 0, 0, nullptr, nullptr);
    gemm(tmpB, 128, 2, Mn, 128, E0, 128, 128, peb, nullptr, 1, 0, 0, 0, 0, nullptr, nullptr);

    // relation inputs + encoder: 28->128->128->128 (relu)
    rel_inputs_kernel<<<(Mr + 255) / 256, 256, 0, stream>>>(
        idxR, idxS, pin, attrs, state, p_instance, relin);
    gemm(relin, 32, 3, Mr,  32, nullptr, 0, 0, tmpA,   nullptr, 1, 0, 0, 0, 0, nullptr, nullptr);
    gemm(tmpA, 128, 4, Mr, 128, nullptr, 0, 0, tmpB,   nullptr, 1, 0, 0, 0, 0, nullptr, nullptr);
    gemm(tmpB, 128, 5, Mr, 128, nullptr, 0, 0, relenc, nullptr, 1, 0, 0, 0, 0, nullptr, nullptr);

    // propagation: 3 steps
    const u16* effb = peb;   // bf16 eff (gather source)
    float*     effF = E0;    // f32 eff (residual source)
    u16* ebping[2] = { eb0, eb1 };
    for (int s = 0; s < 3; ++s) {
        zero_kernel<<<(Mn * 128 + 255) / 256, 256, 0, stream>>>(agg, Mn * 128);
        // effect_rel = relu(W_rp [relenc|e_r|e_s] + b), scattered into agg via idxR
        gemm(nullptr, 0, 6, Mr, 384, nullptr, 0, 0, nullptr, nullptr, 1,
             relenc, effb, idxR, idxS, agg, idxR);
        concat_pp_kernel<<<(Mn * 256 + 255) / 256, 256, 0, stream>>>(peb, agg, cpp);
        float* Enext = (effF == E0) ? E1 : E0;
        u16*   bnext = ebping[s & 1];
        gemm(cpp, 256, 7, Mn, 256, Enext, 128, 128, bnext, effF, 1,
             0, 0, 0, 0, nullptr, nullptr);
        effF = Enext; effb = bnext;
    }

    // decoder on particles: 128->128->128->3
    extract_h_kernel<<<(Mp * 128 + 255) / 256, 256, 0, stream>>>(effb, tmpA);
    gemm(tmpA, 128, 8, Mp, 128, nullptr, 0, 0, tmpB, nullptr, 1, 0, 0, 0, 0, nullptr, nullptr);
    gemm(tmpB, 128, 9, Mp, 128, nullptr, 0, 0, tmpA, nullptr, 1, 0, 0, 0, 0, nullptr, nullptr);
    gemm(tmpA, 128, 10, Mp, 128, (float*)d_out, 3, 3, nullptr, nullptr, 0,
         0, 0, 0, 0, nullptr, nullptr);
}
